// SingleStreamBlockWithText_35064113004848
// MI455X (gfx1250) — compile-verified, matrix operand order checked by reading, not run on hardware
//
#include <hip/hip_runtime.h>
#include <cstddef>
#include <cstdint>

#define B_   2
#define L_   2048
#define T_   512
#define H_   2048
#define NH_  16
#define D_   128
#define MLP_ 8192

typedef __attribute__((ext_vector_type(8)))  float  v8f;
typedef __attribute__((ext_vector_type(16))) __bf16 v16bf;
typedef __attribute__((ext_vector_type(4)))  unsigned int u32x4;
typedef __attribute__((ext_vector_type(8)))  int i32x8;
typedef __attribute__((ext_vector_type(4)))  int i32x4;

union FragBF { uint4 u[2]; v16bf v; };

// ---------------------------------------------------------------------------
// Tensor Data Mover: 2D bf16 tile Global -> LDS with optional row padding.
// D# per CDNA5 ISA. Must be called wave-uniformly.
//   pad codes: interval bytes = 8 << pad_int_code ; amount DWORDs = pad_amt_code+1
// 6-arg builtin on this toolchain (g0, g1, g2, g3, g4, cpol).
// ---------------------------------------------------------------------------
__device__ __forceinline__ void tdm_load_2d_bf16(unsigned lds_off, const void* gp,
                                                 unsigned tile_w, unsigned tile_h,
                                                 unsigned long long stride_elems,
                                                 unsigned pad_int_code,
                                                 unsigned pad_amt_code,
                                                 unsigned pad_en) {
    unsigned long long ga = (unsigned long long)(uintptr_t)gp;
    u32x4 g0;
    g0[0] = 1u;                                              // count=1, user mode
    g0[1] = lds_off;                                         // lds_addr
    g0[2] = (unsigned)(ga & 0xffffffffull);                  // global_addr[31:0]
    g0[3] = (unsigned)((ga >> 32) & 0x1ffffffull)            // global_addr[56:32]
          | 0x80000000u;                                     // type=2 ("image")
    i32x8 g1;
    g1[0] = (int)((1u << 16)                                 // data_size=1 (2 bytes)
                | (pad_en << 20)
                | ((pad_int_code & 7u) << 22)
                | ((pad_amt_code & 127u) << 25));
    g1[1] = (int)((tile_w & 0xffffu) << 16);                 // tensor_dim0[15:0]
    g1[2] = (int)(((tile_w >> 16) & 0xffffu) | ((tile_h & 0xffffu) << 16));
    g1[3] = (int)(((tile_h >> 16) & 0xffffu) | ((tile_w & 0xffffu) << 16)); // tile_dim0
    g1[4] = (int)(tile_h & 0xffffu);                         // tile_dim1, tile_dim2=0
    g1[5] = (int)(stride_elems & 0xffffffffull);             // dim0_stride[31:0]
    g1[6] = (int)((stride_elems >> 32) & 0xffffull);         // dim0_stride[47:32]
    g1[7] = 0;
    i32x4 z4 = {0, 0, 0, 0};
    i32x8 z8 = {0, 0, 0, 0, 0, 0, 0, 0};
    __builtin_amdgcn_tensor_load_to_lds(g0, g1, z4, z4, z8, 0);
}

// ---------------------------------------------------------------------------
// Elementwise / small kernels
// ---------------------------------------------------------------------------
__global__ void k_silu(const float* __restrict__ in, float* __restrict__ out, int n) {
    int i = blockIdx.x * blockDim.x + threadIdx.x;
    if (i < n) { float v = in[i]; out[i] = v * (1.0f / (1.0f + __expf(-v))); }
}

__global__ void k_f2bf(const float* __restrict__ in, __bf16* __restrict__ out, int n) {
    int i = blockIdx.x * blockDim.x + threadIdx.x;
    if (i < n) out[i] = (__bf16)in[i];
}

// one block per output scalar: out[b*OD+o] = dot(sv[b,:], w[o,:]) + bias[o]
__global__ __launch_bounds__(256) void k_rowdot(const float* __restrict__ sv,
                                                const float* __restrict__ w,
                                                const float* __restrict__ bias,
                                                float* __restrict__ out,
                                                int OD, int K) {
    int blk = blockIdx.x;
    int b = blk / OD, o = blk % OD;
    const float* s  = sv + (size_t)b * K;
    const float* wr = w  + (size_t)o * K;
    __shared__ float red[256];
    float p = 0.f;
    for (int c = threadIdx.x; c < K; c += 256) p += s[c] * wr[c];
    red[threadIdx.x] = p; __syncthreads();
    for (int st = 128; st > 0; st >>= 1) {
        if (threadIdx.x < st) red[threadIdx.x] += red[threadIdx.x + st];
        __syncthreads();
    }
    if (threadIdx.x == 0) out[blk] = red[0] + bias[o];
}

// LayerNorm + two modulations, bf16 outputs. one block per (b,l) row.
__global__ __launch_bounds__(256) void k_ln_mod(const float* __restrict__ x,
                                                const float* __restrict__ modb,   // (B,3H)
                                                const float* __restrict__ tmodb,  // (B,2H)
                                                __bf16* __restrict__ xmod,
                                                __bf16* __restrict__ xt) {
    int row = blockIdx.x;
    int b = row / L_;
    const float* xr = x + (size_t)row * H_;
    __shared__ float r1[256], r2[256];
    float s = 0.f, s2 = 0.f;
    for (int c = threadIdx.x; c < H_; c += 256) { float v = xr[c]; s += v; s2 += v * v; }
    r1[threadIdx.x] = s; r2[threadIdx.x] = s2; __syncthreads();
    for (int st = 128; st > 0; st >>= 1) {
        if (threadIdx.x < st) { r1[threadIdx.x] += r1[threadIdx.x + st]; r2[threadIdx.x] += r2[threadIdx.x + st]; }
        __syncthreads();
    }
    float mean = r1[0] / H_;
    float var  = r2[0] / H_ - mean * mean;
    float rstd = rsqrtf(var + 1e-6f);
    const float* shift = modb  + (size_t)b * 3 * H_;
    const float* scale = shift + H_;
    const float* tsc   = tmodb + (size_t)b * 2 * H_;
    const float* tsh   = tsc + H_;
    for (int c = threadIdx.x; c < H_; c += 256) {
        float xn = (xr[c] - mean) * rstd;
        xmod[(size_t)row * H_ + c] = (__bf16)(xn * (1.0f + scale[c]) + shift[c]);
        xt  [(size_t)row * H_ + c] = (__bf16)(xn * (1.0f + tsc[c])   + tsh[c]);
    }
}

// RMS-norm + RoPE for q,k from packed qkv (B,L,3H) bf16.
__global__ __launch_bounds__(64) void k_rmsrope(const __bf16* __restrict__ qkv,
                                                const float*  __restrict__ pe,
                                                const float*  __restrict__ qs,
                                                const float*  __restrict__ ks,
                                                __bf16* __restrict__ qo,
                                                __bf16* __restrict__ ko) {
    int idx = blockIdx.x;                 // ((b*L + l)*NH + h)
    int h  = idx % NH_;
    int bl = idx / NH_;
    int b  = bl / L_, l = bl % L_;
    int p  = threadIdx.x;                 // 0..63
    const __bf16* base = qkv + (size_t)bl * 3 * H_ + h * D_;
    float q0 = (float)base[2 * p],      q1 = (float)base[2 * p + 1];
    float k0 = (float)base[H_ + 2 * p], k1 = (float)base[H_ + 2 * p + 1];
    __shared__ float red[64];
    red[p] = q0 * q0 + q1 * q1; __syncthreads();
    for (int st = 32; st > 0; st >>= 1) { if (p < st) red[p] += red[p + st]; __syncthreads(); }
    float rrq = rsqrtf(red[0] / D_ + 1e-6f); __syncthreads();
    red[p] = k0 * k0 + k1 * k1; __syncthreads();
    for (int st = 32; st > 0; st >>= 1) { if (p < st) red[p] += red[p + st]; __syncthreads(); }
    float rrk = rsqrtf(red[0] / D_ + 1e-6f);
    float qn0 = q0 * rrq * qs[2 * p], qn1 = q1 * rrq * qs[2 * p + 1];
    float kn0 = k0 * rrk * ks[2 * p], kn1 = k1 * rrk * ks[2 * p + 1];
    const float* pp = pe + ((((size_t)b * L_ + l) * 64 + p) * 2) * 2;
    float oq0 = pp[0] * qn0 + pp[1] * qn1;
    float oq1 = pp[2] * qn0 + pp[3] * qn1;
    float ok0 = pp[0] * kn0 + pp[1] * kn1;
    float ok1 = pp[2] * kn0 + pp[3] * kn1;
    size_t o = (size_t)bl * H_ + h * D_ + 2 * p;
    qo[o] = (__bf16)oq0; qo[o + 1] = (__bf16)oq1;
    ko[o] = (__bf16)ok0; ko[o + 1] = (__bf16)ok1;
}

// ---------------------------------------------------------------------------
// Generic WMMA GEMM: C[MxN] = A_bf16[MxK] * W_f32[NxK]^T (+bias, epilogue)
// Software-pipelined double buffering: TDM moves A(s+1) and global loads fetch
// W(s+1) into registers while WMMAs run on buffer s.
// mode: 0 bf16; 1 bf16+gelu; 2 bf16 + f32 residual; 3 f32 gated: x + gate*acc
// ---------------------------------------------------------------------------
__global__ __launch_bounds__(256) void k_gemm_wmma(
    const __bf16* __restrict__ A, int lda,
    const float*  __restrict__ W, int ldw,
    const float*  __restrict__ bias,
    int M, int N, int K,
    void* __restrict__ out, int ldc, int mode,
    const float* __restrict__ res, int ldres,
    const float* __restrict__ gate, int gstride, int rowsPerBatch) {

    __shared__ __align__(16) __bf16 As[2][128][40];   // 32-elem rows padded to 40
    __shared__ __align__(16) __bf16 Ws[2][128][40];

    int tid  = threadIdx.x;
    int lane = tid & 31, wid = tid >> 5;
    int wm = wid >> 2;          // 0..1 : 64 rows each
    int wn = wid & 3;           // 0..3 : 32 cols each
    int m0 = blockIdx.y * 128;
    int n0 = blockIdx.x * 128;
    int mlane = lane & 15, kh = lane >> 4;

    unsigned asOff[2];
    asOff[0] = (unsigned)(size_t)(void*)&As[0][0][0];
    asOff[1] = (unsigned)(size_t)(void*)&As[1][0][0];

    v8f acc[4][2] = {};
    float4 f[4];

    // ---- prologue: stage step 0 into buffer 0 ----
    if (wid == 0) {
        tdm_load_2d_bf16(asOff[0], A + (size_t)m0 * lda,
                         32, 128, (unsigned long long)lda, 3, 3, 1);
    }
#pragma unroll
    for (int i = 0; i < 4; ++i) {
        int id = tid + 256 * i, r = id >> 3, c = id & 7;
        f[i] = *(const float4*)(W + (size_t)(n0 + r) * ldw + c * 4);
    }
#pragma unroll
    for (int i = 0; i < 4; ++i) {
        int id = tid + 256 * i, r = id >> 3, c = id & 7;
        __bf16* d = &Ws[0][r][c * 4];
        d[0] = (__bf16)f[i].x; d[1] = (__bf16)f[i].y;
        d[2] = (__bf16)f[i].z; d[3] = (__bf16)f[i].w;
    }
    if (wid == 0) __builtin_amdgcn_s_wait_tensorcnt(0);
    __syncthreads();

    int nsteps = K >> 5;
    for (int step = 0; step < nsteps; ++step) {
        int cur = step & 1, nxt = cur ^ 1;
        bool hasNext = (step + 1) < nsteps;
        int k1 = (step + 1) << 5;

        // ---- issue next tile's data movement (overlaps with compute) ----
        if (hasNext) {
            if (wid == 0) {
                tdm_load_2d_bf16(asOff[nxt], A + (size_t)m0 * lda + k1,
                                 32, 128, (unsigned long long)lda, 3, 3, 1);
            }
#pragma unroll
            for (int i = 0; i < 4; ++i) {
                int id = tid + 256 * i, r = id >> 3, c = id & 7;
                f[i] = *(const float4*)(W + (size_t)(n0 + r) * ldw + k1 + c * 4);
            }
        }

        // ---- WMMA on current buffer ----
        FragBF a[4], bw[2];
#pragma unroll
        for (int mi = 0; mi < 4; ++mi) {
            int r = wm * 64 + mi * 16 + mlane;
            a[mi].u[0] = *(uint4*)&As[cur][r][kh * 8];
            a[mi].u[1] = *(uint4*)&As[cur][r][16 + kh * 8];
        }
#pragma unroll
        for (int ni = 0; ni < 2; ++ni) {
            int c = wn * 32 + ni * 16 + mlane;
            bw[ni].u[0] = *(uint4*)&Ws[cur][c][kh * 8];
            bw[ni].u[1] = *(uint4*)&Ws[cur][c][16 + kh * 8];
        }
#pragma unroll
        for (int mi = 0; mi < 4; ++mi)
#pragma unroll
            for (int ni = 0; ni < 2; ++ni)
                acc[mi][ni] = __builtin_amdgcn_wmma_f32_16x16x32_bf16(
                    false, a[mi].v, false, bw[ni].v, (short)0, acc[mi][ni], false, false);
        __syncthreads();    // all waves done reading 'cur'

        // ---- finish staging 'nxt' ----
        if (hasNext) {
#pragma unroll
            for (int i = 0; i < 4; ++i) {
                int id = tid + 256 * i, r = id >> 3, c = id & 7;
                __bf16* d = &Ws[nxt][r][c * 4];
                d[0] = (__bf16)f[i].x; d[1] = (__bf16)f[i].y;
                d[2] = (__bf16)f[i].z; d[3] = (__bf16)f[i].w;
            }
            if (wid == 0) __builtin_amdgcn_s_wait_tensorcnt(0);
            __syncthreads();
        }
    }

    int hl = lane >> 4, cn = lane & 15;
#pragma unroll
    for (int mi = 0; mi < 4; ++mi)
#pragma unroll
        for (int ni = 0; ni < 2; ++ni)
#pragma unroll
            for (int r = 0; r < 8; ++r) {
                int grow = m0 + wm * 64 + mi * 16 + r + 8 * hl;
                int gcol = n0 + wn * 32 + ni * 16 + cn;
                float v = acc[mi][ni][r] + bias[gcol];
                if (mode == 0) {
                    ((__bf16*)out)[(size_t)grow * ldc + gcol] = (__bf16)v;
                } else if (mode == 1) {
                    float g = 0.5f * v * (1.0f + tanhf(0.7978845608028654f * (v + 0.044715f * v * v * v)));
                    ((__bf16*)out)[(size_t)grow * ldc + gcol] = (__bf16)g;
                } else if (mode == 2) {
                    v += res[(size_t)grow * ldres + gcol];
                    ((__bf16*)out)[(size_t)grow * ldc + gcol] = (__bf16)v;
                } else { // mode 3
                    int bb = grow / rowsPerBatch;
                    float o = res[(size_t)grow * ldres + gcol] +
                              gate[(size_t)bb * gstride + gcol] * v;
                    ((float*)out)[(size_t)grow * ldc + gcol] = o;
                }
            }
}

// ---------------------------------------------------------------------------
// Flash-attention with WMMA. K chunk staged via TDM; V transposed manually.
// Block: 256 thr = 8 waves, 128 queries; key chunks of 32; D=128.
// ---------------------------------------------------------------------------
__global__ __launch_bounds__(256) void k_attn_wmma(
    const __bf16* __restrict__ Q, int ldq,
    const __bf16* __restrict__ Kp, int ldk,
    const __bf16* __restrict__ Vp, int ldv,
    void* __restrict__ Out, int ldo, int outBf16,
    int Lq, int Lk) {

    __shared__ __align__(16) __bf16 Ks[32][136];   // 128-elem rows padded to 136
    __shared__ __align__(16) __bf16 Vt[128][40];
    __shared__ __align__(16) __bf16 Ps[8][16][40];

    int tid  = threadIdx.x;
    int lane = tid & 31, wid = tid >> 5;
    int mlane = lane & 15, kh = lane >> 4, hl = kh;
    int bh = blockIdx.y;
    int b = bh / NH_, h = bh % NH_;
    int q0 = blockIdx.x * 128;

    const __bf16* Qb = Q  + (size_t)b * Lq * ldq + h * D_;
    const __bf16* Kb = Kp + (size_t)b * Lk * ldk + h * D_;
    const __bf16* Vb = Vp + (size_t)b * Lk * ldv + h * D_;

    unsigned ksOff = (unsigned)(size_t)(void*)&Ks[0][0];

    // Q A-fragments (16 rows x 128 K, split into 4 chunks of 32)
    FragBF aQ[4];
    {
        int qrow = q0 + wid * 16 + mlane;
        const __bf16* qptr = Qb + (size_t)qrow * ldq;
#pragma unroll
        for (int kc = 0; kc < 4; ++kc) {
            aQ[kc].u[0] = *(const uint4*)(qptr + kc * 32 + kh * 8);
            aQ[kc].u[1] = *(const uint4*)(qptr + kc * 32 + 16 + kh * 8);
        }
    }

    float Mr[8], Lr[8];
    v8f O[8] = {};
#pragma unroll
    for (int r = 0; r < 8; ++r) { Mr[r] = -1e30f; Lr[r] = 0.f; }
    const float sc = 0.08838834764831845f;  // 1/sqrt(128)

    for (int kb = 0; kb < Lk; kb += 32) {
        // K chunk 32x128 via TDM (pad 64 DWORDs -> +4 DWORDs)
        if (wid == 0) {
            tdm_load_2d_bf16(ksOff, Kb + (size_t)kb * ldk,
                             /*tile_w=*/128, /*tile_h=*/32,
                             (unsigned long long)ldk,
                             /*pad_int=*/5, /*pad_amt=*/3, /*pad_en=*/1);
        }
        // stage V chunk transposed (128 x 32): batch the global loads first
        {
            int r0 = tid >> 4,         c0 = tid & 15;
            int r1 = (tid + 256) >> 4, c1 = tid & 15;
            uint4 u0 = *(const uint4*)(Vb + (size_t)(kb + r0) * ldv + c0 * 8);
            uint4 u1 = *(const uint4*)(Vb + (size_t)(kb + r1) * ldv + c1 * 8);
            const __bf16* e0 = (const __bf16*)&u0;
            const __bf16* e1 = (const __bf16*)&u1;
#pragma unroll
            for (int j = 0; j < 8; ++j) Vt[c0 * 8 + j][r0] = e0[j];
#pragma unroll
            for (int j = 0; j < 8; ++j) Vt[c1 * 8 + j][r1] = e1[j];
        }
        if (kb + 32 < Lk) {
            __builtin_prefetch(Vb + (size_t)(kb + 32 + (tid & 31)) * ldv + (tid >> 5) * 32, 0, 0);
        }
        if (wid == 0) __builtin_amdgcn_s_wait_tensorcnt(0);
        __syncthreads();

        // S = Q * K^T
        v8f s[2] = {};
#pragma unroll
        for (int j = 0; j < 2; ++j)
#pragma unroll
            for (int kc = 0; kc < 4; ++kc) {
                FragBF bK;
                int key = j * 16 + mlane;
                bK.u[0] = *(uint4*)&Ks[key][kc * 32 + kh * 8];
                bK.u[1] = *(uint4*)&Ks[key][kc * 32 + 16 + kh * 8];
                s[j] = __builtin_amdgcn_wmma_f32_16x16x32_bf16(
                    false, aQ[kc].v, false, bK.v, (short)0, s[j], false, false);
            }

        // online softmax (rows live across 16-lane halves)
#pragma unroll
        for (int r = 0; r < 8; ++r) {
            float v0 = s[0][r] * sc, v1 = s[1][r] * sc;
            float rm = fmaxf(v0, v1);
#pragma unroll
            for (int off = 8; off >= 1; off >>= 1) rm = fmaxf(rm, __shfl_xor(rm, off, 16));
            float newM  = fmaxf(Mr[r], rm);
            float alpha = __expf(Mr[r] - newM);
            float p0 = __expf(v0 - newM), p1 = __expf(v1 - newM);
            float ps = p0 + p1;
#pragma unroll
            for (int off = 8; off >= 1; off >>= 1) ps += __shfl_xor(ps, off, 16);
            Lr[r] = Lr[r] * alpha + ps;
            Mr[r] = newM;
#pragma unroll
            for (int t = 0; t < 8; ++t) O[t][r] *= alpha;
            int mrow = r + 8 * hl;
            Ps[wid][mrow][mlane]      = (__bf16)p0;
            Ps[wid][mrow][16 + mlane] = (__bf16)p1;
        }

        // repack P as A-fragment (same-wave LDS bounce)
        FragBF aP;
        aP.u[0] = *(uint4*)&Ps[wid][mlane][kh * 8];
        aP.u[1] = *(uint4*)&Ps[wid][mlane][16 + kh * 8];

        // O += P * V
#pragma unroll
        for (int t = 0; t < 8; ++t) {
            FragBF bV;
            int vr = t * 16 + mlane;
            bV.u[0] = *(uint4*)&Vt[vr][kh * 8];
            bV.u[1] = *(uint4*)&Vt[vr][16 + kh * 8];
            O[t] = __builtin_amdgcn_wmma_f32_16x16x32_bf16(
                false, aP.v, false, bV.v, (short)0, O[t], false, false);
        }
        __syncthreads();
    }

    // epilogue: normalize and store
#pragma unroll
    for (int r = 0; r < 8; ++r) {
        float inv = 1.0f / Lr[r];
        int grow = q0 + wid * 16 + r + 8 * hl;
        size_t rowoff = (size_t)(b * Lq + grow) * ldo + h * D_;
#pragma unroll
        for (int t = 0; t < 8; ++t) {
            float v = O[t][r] * inv;
            int gcol = t * 16 + mlane;
            if (outBf16) ((__bf16*)Out)[rowoff + gcol] = (__bf16)v;
            else         ((float*)Out)[rowoff + gcol]  = v;
        }
    }
}

// ---------------------------------------------------------------------------
// Host launcher
// ---------------------------------------------------------------------------
extern "C" void kernel_launch(void* const* d_in, const int* in_sizes, int n_in,
                              void* d_out, int out_size, void* d_ws, size_t ws_size,
                              hipStream_t stream) {
    const float* x      = (const float*)d_in[0];
    const float* vec    = (const float*)d_in[1];
    const float* pe     = (const float*)d_in[2];
    const float* text   = (const float*)d_in[3];
    const float* lin1_w = (const float*)d_in[4];
    const float* lin1_b = (const float*)d_in[5];
    const float* lin2_w = (const float*)d_in[6];
    const float* lin2_b = (const float*)d_in[7];
    const float* qnorm  = (const float*)d_in[8];
    const float* knorm  = (const float*)d_in[9];
    const float* mod_w  = (const float*)d_in[10];
    const float* mod_b  = (const float*)d_in[11];
    const float* tmod_w = (const float*)d_in[12];
    const float* tmod_b = (const float*)d_in[13];
    const float* tin_w  = (const float*)d_in[14];
    const float* tin_b  = (const float*)d_in[15];
    const float* tout_w = (const float*)d_in[16];
    const float* tout_b = (const float*)d_in[17];

    const int BL = B_ * L_;         // 4096
    const int BT = B_ * T_;         // 1024

    char* w = (char*)d_ws;
    size_t off = 0;
    auto alloc = [&](size_t bytes) -> void* {
        void* p = w + off;
        off = (off + bytes + 255) & ~(size_t)255;
        return p;
    };

    float*  sv     = (float*)alloc((size_t)B_ * H_ * 4);
    float*  modb   = (float*)alloc((size_t)B_ * 3 * H_ * 4);
    float*  tmodb  = (float*)alloc((size_t)B_ * 2 * H_ * 4);
    __bf16* xmod   = (__bf16*)alloc((size_t)BL * H_ * 2);
    __bf16* xt     = (__bf16*)alloc((size_t)BL * H_ * 2);
    __bf16* textbf = (__bf16*)alloc((size_t)BT * H_ * 2);
    __bf16* qkv    = (__bf16*)alloc((size_t)BL * 3 * H_ * 2);
    __bf16* qbf    = (__bf16*)alloc((size_t)BL * H_ * 2);
    __bf16* kbf    = (__bf16*)alloc((size_t)BL * H_ * 2);
    __bf16* tq     = (__bf16*)alloc((size_t)BL * H_ * 2);
    __bf16* tk     = (__bf16*)alloc((size_t)BT * H_ * 2);
    __bf16* tv     = (__bf16*)alloc((size_t)BT * H_ * 2);
    float*  xattn  = (float*)alloc((size_t)BL * H_ * 4);
    __bf16* xtext  = (__bf16*)alloc((size_t)BL * H_ * 2);
    __bf16* cat    = (__bf16*)alloc((size_t)BL * (H_ + MLP_) * 2);

    // 1) vector path
    k_silu<<<(B_ * H_ + 255) / 256, 256, 0, stream>>>(vec, sv, B_ * H_);
    k_rowdot<<<B_ * 3 * H_, 256, 0, stream>>>(sv, mod_w, mod_b, modb, 3 * H_, H_);
    k_rowdot<<<B_ * 2 * H_, 256, 0, stream>>>(sv, tmod_w, tmod_b, tmodb, 2 * H_, H_);

    // 2) LN + modulation -> bf16
    k_ln_mod<<<BL, 256, 0, stream>>>(x, modb, tmodb, xmod, xt);
    k_f2bf<<<(BT * H_ + 255) / 256, 256, 0, stream>>>(text, textbf, BT * H_);

    // 3) lin1: qkv part (mode 0) and mlp part (mode 1, gelu -> cat[:, H:])
    {
        dim3 g(6144 / 128, BL / 128);
        k_gemm_wmma<<<g, 256, 0, stream>>>(xmod, H_, lin1_w, H_, lin1_b,
                                           BL, 3 * H_, H_, qkv, 3 * H_, 0,
                                           nullptr, 0, nullptr, 0, L_);
    }
    {
        dim3 g(MLP_ / 128, BL / 128);
        k_gemm_wmma<<<g, 256, 0, stream>>>(xmod, H_, lin1_w + (size_t)3 * H_ * H_, H_,
                                           lin1_b + 3 * H_,
                                           BL, MLP_, H_, cat + H_, H_ + MLP_, 1,
                                           nullptr, 0, nullptr, 0, L_);
    }

    // 4) rms + rope on q,k
    k_rmsrope<<<BL * NH_, 64, 0, stream>>>(qkv, pe, qnorm, knorm, qbf, kbf);

    // 5) self attention (V read in place from qkv with ld=3H)
    {
        dim3 g(L_ / 128, B_ * NH_);
        k_attn_wmma<<<g, 256, 0, stream>>>(qbf, H_, kbf, H_, qkv + 2 * H_, 3 * H_,
                                           xattn, H_, 0, L_, L_);
    }

    // 6) text q/k/v projections
    {
        dim3 g(H_ / 128, BL / 128);
        k_gemm_wmma<<<g, 256, 0, stream>>>(xt, H_, tin_w, H_, tin_b,
                                           BL, H_, H_, tq, H_, 0,
                                           nullptr, 0, nullptr, 0, L_);
    }
    {
        dim3 g(H_ / 128, BT / 128);
        k_gemm_wmma<<<g, 256, 0, stream>>>(textbf, H_, tin_w + (size_t)H_ * H_, H_,
                                           tin_b + H_, BT, H_, H_, tk, H_, 0,
                                           nullptr, 0, nullptr, 0, T_);
        k_gemm_wmma<<<g, 256, 0, stream>>>(textbf, H_, tin_w + (size_t)2 * H_ * H_, H_,
                                           tin_b + 2 * H_, BT, H_, H_, tv, H_, 0,
                                           nullptr, 0, nullptr, 0, T_);
    }

    // 7) cross attention -> bf16
    {
        dim3 g(L_ / 128, B_ * NH_);
        k_attn_wmma<<<g, 256, 0, stream>>>(tq, H_, tk, H_, tv, H_,
                                           xtext, H_, 1, L_, T_);
    }

    // 8) tout projection + x_attn residual -> cat[:, :H] (mode 2)
    {
        dim3 g(H_ / 128, BL / 128);
        k_gemm_wmma<<<g, 256, 0, stream>>>(xtext, H_, tout_w, H_, tout_b,
                                           BL, H_, H_, cat, H_ + MLP_, 2,
                                           xattn, H_, nullptr, 0, L_);
    }

    // 9) lin2 + gated residual -> d_out (mode 3)
    {
        dim3 g(H_ / 128, BL / 128);
        k_gemm_wmma<<<g, 256, 0, stream>>>(cat, H_ + MLP_, lin2_w, H_ + MLP_, lin2_b,
                                           BL, H_, H_ + MLP_, d_out, H_, 3,
                                           x, H_, modb + 2 * H_, 3 * H_, L_);
    }
}
